// FSDPAAttnMaskOneshot_78572131713228
// MI455X (gfx1250) — compile-verified, matrix operand order checked by reading, not run on hardware
//
#include <hip/hip_runtime.h>
#include <hip/hip_bf16.h>

typedef __attribute__((ext_vector_type(16))) __bf16 v16bf;
typedef __attribute__((ext_vector_type(8)))  __bf16 v8bf;
typedef __attribute__((ext_vector_type(8)))  float  v8f;
typedef int v4i_vs __attribute__((vector_size(4 * sizeof(int))));

#define S_LEN 2048
#define D_IN  1280
#define E_OUT 3840
#define H_N   16
#define HD_N  80
#define HDP   96          // hd padded to 96 for Q/K fragment loads

// ds_swizzle group-of-32: imm = xor_mask<<10 | or_mask<<5 | and_mask
#define SWIZ_XOR_F(x, m) __int_as_float(__builtin_amdgcn_ds_swizzle(__float_as_int(x), (((m) << 10) | 0x1f)))

__device__ __forceinline__ float half16_max(float x) {
  x = fmaxf(x, SWIZ_XOR_F(x, 1));
  x = fmaxf(x, SWIZ_XOR_F(x, 2));
  x = fmaxf(x, SWIZ_XOR_F(x, 4));
  x = fmaxf(x, SWIZ_XOR_F(x, 8));
  return x;
}
__device__ __forceinline__ float half16_sum(float x) {
  x += SWIZ_XOR_F(x, 1);
  x += SWIZ_XOR_F(x, 2);
  x += SWIZ_XOR_F(x, 4);
  x += SWIZ_XOR_F(x, 8);
  return x;
}

__device__ __forceinline__ v16bf cat8(v8bf a, v8bf b) {
  return __builtin_shufflevector(a, b, 0, 1, 2, 3, 4, 5, 6, 7, 8, 9, 10, 11, 12, 13, 14, 15);
}

// async global -> LDS copy of one 16B chunk (per-lane addresses), ASYNCcnt-tracked
__device__ __forceinline__ void async_copy_b128(const __bf16* g, __bf16* l) {
  __builtin_amdgcn_global_load_async_to_lds_b128(
      (__attribute__((address_space(1))) v4i_vs*)g,
      (__attribute__((address_space(3))) v4i_vs*)l, 0, 0);
}
__device__ __forceinline__ void wait_asynccnt0() {
  asm volatile("s_wait_asynccnt 0x0" ::: "memory");
}

// -------------------------------------------------------------------------
// Kernel 0: zero the hd in [80,96) pad columns of Qp and Kp.
// -------------------------------------------------------------------------
__global__ void __launch_bounds__(256)
pad_zero_kernel(__bf16* __restrict__ qkbase) {
  const int r = blockIdx.x * 256 + threadIdx.x;      // row id over Qp+Kp: 2*H*S rows
  if (r < 2 * H_N * S_LEN) {
    __bf16* p = qkbase + (size_t)r * HDP + HD_N;     // 16 bf16 tail, 16B-aligned
#pragma unroll
    for (int i = 0; i < 16; ++i) p[i] = (__bf16)0.0f;
  }
}

// -------------------------------------------------------------------------
// Kernel 1: qkv[s,e] = sum_d x[s,d]*W[e,d] + b[e], bf16 outputs:
//   Qp,Kp : [H][S][96]  (row-major, zero-padded hd)
//   Vt    : [H][HD][S]  (transposed: hd-major, keys contiguous)
// -------------------------------------------------------------------------
__global__ void __launch_bounds__(256)
qkv_gemm_bf16_wmma(const float* __restrict__ x, const float* __restrict__ W,
                   const float* __restrict__ bias,
                   __bf16* __restrict__ Qp, __bf16* __restrict__ Kp,
                   __bf16* __restrict__ Vt) {
  const int lane  = threadIdx.x & 31;
  const int wave  = threadIdx.x >> 5;
  const int s0    = blockIdx.x * 128 + wave * 16;
  const int e0    = blockIdx.y * 64;
  const int half  = lane >> 4;
  const int idx16 = lane & 15;
  const int kbase = half * 8;

  v8f acc[4] = {v8f{}, v8f{}, v8f{}, v8f{}};

  const float* arow = x + (size_t)(s0 + idx16) * D_IN;

  for (int kk = 0; kk < D_IN; kk += 32) {
    v16bf a;
#pragma unroll
    for (int i = 0; i < 8; ++i) {
      a[i]     = (__bf16)arow[kk + kbase + i];
      a[8 + i] = (__bf16)arow[kk + kbase + 16 + i];
    }
#pragma unroll
    for (int n = 0; n < 4; ++n) {
      const float* brow = W + (size_t)(e0 + n * 16 + idx16) * D_IN;
      v16bf b;
#pragma unroll
      for (int i = 0; i < 8; ++i) {
        b[i]     = (__bf16)brow[kk + kbase + i];
        b[8 + i] = (__bf16)brow[kk + kbase + 16 + i];
      }
      acc[n] = __builtin_amdgcn_wmma_f32_16x16x32_bf16(false, a, false, b,
                                                       (short)0, acc[n], false, false);
    }
  }

#pragma unroll
  for (int n = 0; n < 4; ++n) {
    const int e     = e0 + n * 16 + idx16;
    const float bv  = bias[e];
    const int which = e / D_IN;              // uniform across the 16-lane group
    const int r     = e - which * D_IN;
    const int h     = r / HD_N;              // uniform
    const int hd    = r - h * HD_N;          // = hd0 + idx16
    if (which == 2) {
      // Vt[h][hd][s]: D-layout rows are s-consecutive per lane -> one 16B store
      v8bf pk;
#pragma unroll
      for (int v = 0; v < 8; ++v) pk[v] = (__bf16)(acc[n][v] + bv);
      *(v8bf*)(Vt + ((size_t)(h * HD_N + hd)) * S_LEN + s0 + 8 * half) = pk;
    } else {
      __bf16* base = (which ? Kp : Qp) + ((size_t)h * S_LEN) * HDP + hd;
#pragma unroll
      for (int v = 0; v < 8; ++v) {
        const int srow = s0 + v + 8 * half;
        base[(size_t)srow * HDP] = (__bf16)(acc[n][v] + bv);
      }
    }
  }
}

// -------------------------------------------------------------------------
// Kernel 2: block-diagonal flash attention.
// Block = 128 threads (4 waves) = 64 query rows of one head, sharing one segment.
// K tiles async-staged (double-buffered) into LDS via GLOBAL_LOAD_ASYNC_TO_LDS.
// -------------------------------------------------------------------------
__global__ void __launch_bounds__(128)
attn_flash_bf16_wmma(const __bf16* __restrict__ Qp, const __bf16* __restrict__ Kp,
                     const __bf16* __restrict__ Vt, const int* __restrict__ cu,
                     int nseq, float* __restrict__ out) {
  const int tid   = threadIdx.x;
  const int lane  = tid & 31;
  const int wave  = tid >> 5;                 // 0..3
  const int h     = blockIdx.y;
  const int sblk  = blockIdx.x * 64;
  const int s0    = sblk + wave * 16;
  const int half  = lane >> 4;
  const int idx16 = lane & 15;
  const int kbase = half * 8;
  const float scale = 0.11180339887498949f;   // 1/sqrt(80)

  // block-level segment (blocks assumed not to straddle segment boundaries)
  int seg = 0;
  for (int i = 0; i < nseq; ++i)
    if (cu[i] <= sblk) seg = i;
  const int kv_start = cu[seg];
  const int kv_end   = cu[seg + 1];

  const __bf16* Qh = Qp + (size_t)h * S_LEN * HDP;
  const __bf16* Kh = Kp + (size_t)h * S_LEN * HDP;
  const __bf16* Vh = Vt + (size_t)h * HD_N * S_LEN;

  // Q fragments (16x96 padded), pure 16B loads, no masking needed
  v16bf qf[3];
  const __bf16* qrow = Qh + (size_t)(s0 + idx16) * HDP;
#pragma unroll
  for (int ks = 0; ks < 3; ++ks)
    qf[ks] = cat8(*(const v8bf*)(qrow + ks * 32 + kbase),
                  *(const v8bf*)(qrow + ks * 32 + kbase + 16));

  float m[8], l[8];
  v8f acc[5] = {v8f{}, v8f{}, v8f{}, v8f{}, v8f{}};
#pragma unroll
  for (int v = 0; v < 8; ++v) { m[v] = -3.0e38f; l[v] = 0.0f; }

  __shared__ __align__(16) __bf16 ksmem[2][32][HDP];  // double-buffered K tile: 12KB
  __shared__ __align__(16) __bf16 pbuf[4][16][32];    // per-wave P tiles: 4KB

  // async stage of K tile kb into buffer b: 384 x 16B chunks over 128 threads
  auto stage = [&](int b, int kb) {
#pragma unroll
    for (int j = 0; j < 3; ++j) {
      const int c    = tid + 128 * j;        // 0..383
      const int row  = c / 12;               // 12 chunks per 192B row
      const int col8 = (c - row * 12) * 8;
      int key = kb + row;
      if (key > S_LEN - 1) key = S_LEN - 1;  // clamp; masked later via -inf
      async_copy_b128(Kh + (size_t)key * HDP + col8, &ksmem[b][row][col8]);
    }
  };

  stage(0, kv_start);
  wait_asynccnt0();
  __syncthreads();

  int buf = 0;
  for (int kb = kv_start; kb < kv_end; kb += 32) {
    // prefetch next K tile into the other buffer (uniform condition)
    if (kb + 32 < kv_end) stage(buf ^ 1, kb + 32);

    // --- scores for two 16-key subtiles ---
    v8f sc[2];
#pragma unroll
    for (int sub = 0; sub < 2; ++sub) {
      const int lrow = sub * 16 + idx16;
      v8f c = {};
#pragma unroll
      for (int ks = 0; ks < 3; ++ks) {
        v16bf bfr = cat8(*(const v8bf*)&ksmem[buf][lrow][ks * 32 + kbase],
                         *(const v8bf*)&ksmem[buf][lrow][ks * 32 + kbase + 16]);
        c = __builtin_amdgcn_wmma_f32_16x16x32_bf16(false, qf[ks], false, bfr,
                                                    (short)0, c, false, false);
      }
      const bool kvalid = (kb + sub * 16 + idx16) < kv_end;
#pragma unroll
      for (int v = 0; v < 8; ++v)
        sc[sub][v] = kvalid ? c[v] * scale : -3.0e38f;
    }

    // --- online softmax (rows live per 16-lane half) ---
    float mnew[8], cf[8];
#pragma unroll
    for (int v = 0; v < 8; ++v) {
      float mx = fmaxf(sc[0][v], sc[1][v]);
      mx = half16_max(mx);
      mnew[v] = fmaxf(m[v], mx);
      cf[v]   = __expf(m[v] - mnew[v]);
      m[v]    = mnew[v];
    }
#pragma unroll
    for (int t = 0; t < 5; ++t)
#pragma unroll
      for (int v = 0; v < 8; ++v) acc[t][v] *= cf[v];

    // --- P = exp(sc - m) -> LDS for A-fragment reshape ---
    float psum[8];
#pragma unroll
    for (int v = 0; v < 8; ++v) psum[v] = 0.0f;
#pragma unroll
    for (int sub = 0; sub < 2; ++sub) {
#pragma unroll
      for (int v = 0; v < 8; ++v) {
        const float p = __expf(sc[sub][v] - m[v]);   // masked keys -> exactly 0
        psum[v] += p;
        pbuf[wave][v + 8 * half][sub * 16 + idx16] = (__bf16)p;
      }
    }
#pragma unroll
    for (int v = 0; v < 8; ++v)
      l[v] = l[v] * cf[v] + half16_sum(psum[v]);

    __builtin_amdgcn_wave_barrier();   // same-wave DS in-order; block compiler reordering

    v16bf pf = cat8(*(const v8bf*)&pbuf[wave][idx16][kbase],
                    *(const v8bf*)&pbuf[wave][idx16][kbase + 16]);
    __builtin_amdgcn_wave_barrier();

    // --- acc += P x V: Vt rows are key-contiguous -> two 16B loads per tile ---
    int b0 = kb + kbase;      if (b0 > S_LEN - 8) b0 = S_LEN - 8;
    int b1 = kb + kbase + 16; if (b1 > S_LEN - 8) b1 = S_LEN - 8;
#pragma unroll
    for (int t = 0; t < 5; ++t) {
      const __bf16* vrow = Vh + (size_t)(t * 16 + idx16) * S_LEN;
      v16bf vf = cat8(*(const v8bf*)(vrow + b0), *(const v8bf*)(vrow + b1));
      acc[t] = __builtin_amdgcn_wmma_f32_16x16x32_bf16(false, pf, false, vf,
                                                       (short)0, acc[t], false, false);
    }

    // prefetched tile complete + all waves done with current buffer
    wait_asynccnt0();
    __syncthreads();
    buf ^= 1;
  }

  // --- normalize and store out[b=0][s][h][hd] (f32) ---
  float invl[8];
#pragma unroll
  for (int v = 0; v < 8; ++v) invl[v] = 1.0f / l[v];
#pragma unroll
  for (int t = 0; t < 5; ++t) {
#pragma unroll
    for (int v = 0; v < 8; ++v) {
      const int s = s0 + v + 8 * half;
      out[(size_t)s * (H_N * HD_N) + h * HD_N + t * 16 + idx16] = acc[t][v] * invl[v];
    }
  }
}

extern "C" void kernel_launch(void* const* d_in, const int* in_sizes, int n_in,
                              void* d_out, int out_size, void* d_ws, size_t ws_size,
                              hipStream_t stream) {
  const float* x  = (const float*)d_in[0];   // [S,1,D]
  const int*   cu = (const int*)d_in[1];     // [NSEQ+1]
  const float* W  = (const float*)d_in[2];   // [3D, D]
  const float* b  = (const float*)d_in[3];   // [3D]
  float* out = (float*)d_out;                // [1,S,H,HD] f32

  // bf16 workspace: Qp [H][S][96] | Kp [H][S][96] | Vt [H][80][S]  (~17.8 MB)
  __bf16* ws0 = (__bf16*)d_ws;
  const size_t QK_SZ = (size_t)H_N * S_LEN * HDP;
  __bf16* Qp = ws0;
  __bf16* Kp = ws0 + QK_SZ;
  __bf16* Vt = ws0 + 2 * QK_SZ;

  const int nseq = in_sizes[1] - 1;

  // zero the hd pad columns of Qp/Kp (Qp,Kp contiguous: 2*H*S rows)
  const int nrows = 2 * H_N * S_LEN;
  pad_zero_kernel<<<(nrows + 255) / 256, 256, 0, stream>>>(Qp);

  dim3 g1(S_LEN / 128, E_OUT / 64);          // 16 x 60 blocks, 8 waves each
  qkv_gemm_bf16_wmma<<<g1, 256, 0, stream>>>(x, W, b, Qp, Kp, Vt);

  dim3 g2(S_LEN / 64, H_N);                  // 32 x 16 blocks, 4 waves each
  attn_flash_bf16_wmma<<<g2, 128, 0, stream>>>(Qp, Kp, Vt, cu, nseq, out);
}
